// SpatialTransformerBlock_41884521070882
// MI455X (gfx1250) — matrix operand order repaired and checked by reading, not compiled, not run on hardware
//
#include <hip/hip_runtime.h>
#include <hip/hip_bf16.h>
#include <math.h>

// ---------------------------------------------------------------------------
// Problem constants (from reference setup_inputs)
// ---------------------------------------------------------------------------
#define BB   2
#define NN   4096     // tokens
#define CC   320      // channels
#define HH   8        // heads
#define DHH  40       // head dim
#define LC   77       // context length
#define LCP  160      // context rows padded to multiple of 16 (BB*LC=154 -> 160)
#define DC   768      // context dim
#define NA   3072     // non-dst tokens
#define ND   1024     // dst tokens
#define RR   2048     // merged src count r
#define NM   2048     // merged token count (unm 1024 + dst 1024)
#define NFF1 2560     // ff_w1 output (2*dff)
#define DFF  1280     // dff

typedef _Float16 h8_t   __attribute__((ext_vector_type(8)));
typedef _Float16 v16h_t __attribute__((ext_vector_type(16)));
typedef float    v8f_t  __attribute__((ext_vector_type(8)));

// ---------------------------------------------------------------------------
// WMMA GEMM: C[M,N] = A[M,K] * Bt[N,K]^T (+bias[N]) (+residual[M,N])
// A, Bt f16 row-major. REQUIRES: M % 16 == 0, N % 32 == 0, K % 32 == 0.
// One wave computes a 16x32 tile: two v_wmma_f32_16x16x32_f16 per K-step
// sharing one A fragment (halves A traffic vs one-tile-per-wave).
//
// A-operand lane layout used here (16-bit A 16x32; contiguous k, not the ISA table order):
//   lane L: M = L%16, g = L/16; halves h: K = k0+g*16+h
// B-operand (32x16): lane L: N = L%16, g = L/16; halves h: K = k0+g*16+h
// C/D (16x16 f32): lane L: N = L%16; acc[r]: M = r + 8*(L/16)
// ---------------------------------------------------------------------------
__global__ void __launch_bounds__(256)
wmma_gemm_kernel(const _Float16* __restrict__ A, const _Float16* __restrict__ Bt,
                 float* __restrict__ C, const float* __restrict__ bias,
                 const float* __restrict__ residual,
                 int M, int N, int K)
{
    const int lane = threadIdx.x & 31;
    const int wid  = threadIdx.x >> 5;
    const int ntn  = N >> 5;                 // 32-wide N tiles
    const int ntm  = M >> 4;
    const int tile = blockIdx.x * 8 + wid;
    if (tile >= ntm * ntn) return;           // wave-uniform exit
    const int tm   = tile / ntn;
    const int tn   = tile % ntn;
    const int g    = lane >> 4;
    const int l16  = lane & 15;
    const int row  = tm * 16 + l16;          // A row fed by this lane
    const int col0 = tn * 32 + l16;          // B col fed by this lane (acc0)
    const int col1 = col0 + 16;              // (acc1)

    const _Float16* ap  = A  + (size_t)row  * K + g * 16;
    const _Float16* bp0 = Bt + (size_t)col0 * K + g * 16;
    const _Float16* bp1 = Bt + (size_t)col1 * K + g * 16;

    v8f_t acc0 = {};
    v8f_t acc1 = {};
    for (int k0 = 0; k0 < K; k0 += 32) {
        h8_t alo = *(const h8_t*)(ap);
        h8_t ahi = *(const h8_t*)(ap + 8);
        h8_t b0l = *(const h8_t*)(bp0);
        h8_t b0h = *(const h8_t*)(bp0 + 8);
        h8_t b1l = *(const h8_t*)(bp1);
        h8_t b1h = *(const h8_t*)(bp1 + 8);
        v16h_t a  = __builtin_shufflevector(alo, ahi, 0,1,2,3,4,5,6,7,8,9,10,11,12,13,14,15);
        v16h_t b0 = __builtin_shufflevector(b0l, b0h, 0,1,2,3,4,5,6,7,8,9,10,11,12,13,14,15);
        v16h_t b1 = __builtin_shufflevector(b1l, b1h, 0,1,2,3,4,5,6,7,8,9,10,11,12,13,14,15);
        acc0 = __builtin_amdgcn_wmma_f32_16x16x32_f16(
            false, a, false, b0, (short)0, acc0, false, false);
        acc1 = __builtin_amdgcn_wmma_f32_16x16x32_f16(
            false, a, false, b1, (short)0, acc1, false, false);
        ap += 32; bp0 += 32; bp1 += 32;
    }
    const float bias0 = bias ? bias[col0] : 0.f;
    const float bias1 = bias ? bias[col1] : 0.f;
#pragma unroll
    for (int r = 0; r < 8; ++r) {
        size_t m = (size_t)(tm * 16 + r + 8 * g);
        float v0 = acc0[r] + bias0;
        float v1 = acc1[r] + bias1;
        if (residual) {
            v0 += residual[m * N + col0];
            v1 += residual[m * N + col1];
        }
        C[m * N + col0] = v0;
        C[m * N + col1] = v1;
    }
}

// ---------------------------------------------------------------------------
// ToMe index construction: dst = (row%2==0 && col%2==0) on a 64x64 grid.
// b_idx = dst positions in row-major order; a_idx = the rest.
// ---------------------------------------------------------------------------
__global__ void build_idx_kernel(int* __restrict__ a_idx, int* __restrict__ b_idx)
{
    int p = blockIdx.x * blockDim.x + threadIdx.x;
    if (p >= NN) return;
    int r = p >> 6, c = p & 63;
    bool is_dst = ((r & 1) == 0) && ((c & 1) == 0);
    int nd_before = ((r + 1) >> 1) * 32 + (((r & 1) == 0) ? ((c + 1) >> 1) : 0);
    if (is_dst) b_idx[nd_before]     = p;
    else        a_idx[p - nd_before] = p;
}

// f32 [K,N] -> f16 [N,K] (transpose + downconvert)
__global__ void transpose_f32_f16_kernel(const float* __restrict__ src,
                                         _Float16* __restrict__ dst, int K, int N)
{
    size_t t = (size_t)blockIdx.x * blockDim.x + threadIdx.x;
    if (t >= (size_t)K * N) return;
    int k = (int)(t / N), n = (int)(t % N);
    dst[(size_t)n * K + k] = (_Float16)src[t];
}

__global__ void f32_to_f16_kernel(const float* __restrict__ in,
                                  _Float16* __restrict__ out, size_t n)
{
    size_t t = (size_t)blockIdx.x * blockDim.x + threadIdx.x;
    if (t < n) out[t] = (_Float16)in[t];
}

// L2-normalize gathered token rows of net: out[b,i,:] = net[b, idx[i], :]/||.||
__global__ void __launch_bounds__(256)
normalize_gather_kernel(const float* __restrict__ net, const int* __restrict__ idx,
                        _Float16* __restrict__ out, int rowsPerBatch)
{
    int w    = (int)((blockIdx.x * (size_t)blockDim.x + threadIdx.x) >> 5);
    int lane = threadIdx.x & 31;
    if (w >= BB * rowsPerBatch) return;
    int b = w / rowsPerBatch, i = w % rowsPerBatch;
    const float* xp = net + ((size_t)b * NN + idx[i]) * CC;
    float v[10];
#pragma unroll
    for (int j = 0; j < 10; ++j) v[j] = xp[lane + 32 * j];
    float s = 0.f;
#pragma unroll
    for (int j = 0; j < 10; ++j) s += v[j] * v[j];
    for (int off = 16; off; off >>= 1) s += __shfl_xor(s, off, 32);
    float inv = rsqrtf(s);
    _Float16* op = out + (size_t)w * CC;
#pragma unroll
    for (int j = 0; j < 10; ++j) op[lane + 32 * j] = (_Float16)(v[j] * inv);
}

// Row max + first-argmax over scores (B*NA rows, ND cols)
__global__ void __launch_bounds__(256)
rowmax_argmax_kernel(const float* __restrict__ scores,
                     float* __restrict__ node_max, int* __restrict__ node_idx)
{
    int w    = (int)((blockIdx.x * (size_t)blockDim.x + threadIdx.x) >> 5);
    int lane = threadIdx.x & 31;
    if (w >= BB * NA) return;
    const float* sp = scores + (size_t)w * ND;
    float best = -3.4e38f; int bidx = 0;
    for (int j = lane; j < ND; j += 32) {
        float v = sp[j];
        if (v > best) { best = v; bidx = j; }
    }
    for (int off = 16; off; off >>= 1) {
        float ov = __shfl_xor(best, off, 32);
        int   oi = __shfl_xor(bidx, off, 32);
        if (ov > best || (ov == best && oi < bidx)) { best = ov; bidx = oi; }
    }
    if (lane == 0) { node_max[w] = best; node_idx[w] = bidx; }
}

// Stable descending rank sort: edge_idx[rank] = i
__global__ void rank_kernel(const float* __restrict__ node_max,
                            int* __restrict__ edge_idx)
{
    int t = blockIdx.x * blockDim.x + threadIdx.x;
    if (t >= BB * NA) return;
    int b = t / NA, i = t % NA;
    const float* vm = node_max + (size_t)b * NA;
    float v = vm[i];
    int cnt = 0;
    for (int j = 0; j < NA; ++j) {
        float u = vm[j];
        cnt += (u > v) || (u == v && j < i);
    }
    edge_idx[(size_t)b * NA + cnt] = i;
}

__global__ void split_idx_kernel(const int* __restrict__ edge_idx,
                                 const int* __restrict__ node_idx,
                                 int* __restrict__ src_idx,
                                 int* __restrict__ unm_idx,
                                 int* __restrict__ dst_idx)
{
    int t = blockIdx.x * blockDim.x + threadIdx.x;
    if (t >= BB * NA) return;
    int b = t / NA, k = t % NA;
    int e = edge_idx[(size_t)b * NA + k];
    if (k < RR) {
        src_idx[(size_t)b * RR + k] = e;
        dst_idx[(size_t)b * RR + k] = node_idx[(size_t)b * NA + e];
    } else {
        unm_idx[(size_t)b * ND + (k - RR)] = e;
    }
}

// LayerNorm over last dim (320), wave per row
__global__ void __launch_bounds__(256)
ln_kernel(const float* __restrict__ x, const float* __restrict__ gamma,
          const float* __restrict__ beta, float* __restrict__ out, int rows)
{
    int w    = (int)((blockIdx.x * (size_t)blockDim.x + threadIdx.x) >> 5);
    int lane = threadIdx.x & 31;
    if (w >= rows) return;
    const float* xp = x + (size_t)w * CC;
    float v[10];
#pragma unroll
    for (int j = 0; j < 10; ++j) v[j] = xp[lane + 32 * j];
    float s = 0.f;
#pragma unroll
    for (int j = 0; j < 10; ++j) s += v[j];
    for (int off = 16; off; off >>= 1) s += __shfl_xor(s, off, 32);
    float mu = s * (1.f / CC);
    float q = 0.f;
#pragma unroll
    for (int j = 0; j < 10; ++j) { float d = v[j] - mu; q += d * d; }
    for (int off = 16; off; off >>= 1) q += __shfl_xor(q, off, 32);
    float rstd = rsqrtf(q * (1.f / CC) + 1e-5f);
    float* op = out + (size_t)w * CC;
#pragma unroll
    for (int j = 0; j < 10; ++j) {
        int c = lane + 32 * j;
        op[c] = (v[j] - mu) * rstd * gamma[c] + beta[c];
    }
}

__global__ void counts_kernel(const int* __restrict__ dst_idx, float* __restrict__ counts)
{
    int t = blockIdx.x * blockDim.x + threadIdx.x;
    if (t >= BB * RR) return;
    int b = t / RR;
    atomicAdd(&counts[(size_t)b * ND + dst_idx[t]], 1.0f);
}

__global__ void merge_sums_kernel(const float* __restrict__ xln,
                                  const int* __restrict__ a_idx,
                                  const int* __restrict__ src_idx,
                                  const int* __restrict__ dst_idx,
                                  float* __restrict__ sums)
{
    size_t t = (size_t)blockIdx.x * blockDim.x + threadIdx.x;
    if (t >= (size_t)BB * RR * CC) return;
    int c = (int)(t % CC);
    size_t r = t / CC;
    int k = (int)(r % RR), b = (int)(r / RR);
    int d   = dst_idx[(size_t)b * RR + k];
    int tok = a_idx[src_idx[(size_t)b * RR + k]];
    atomicAdd(&sums[((size_t)b * ND + d) * CC + c],
              xln[((size_t)b * NN + tok) * CC + c]);
}

__global__ void build_xm_kernel(const float* __restrict__ xln,
                                const int* __restrict__ a_idx,
                                const int* __restrict__ b_idx,
                                const int* __restrict__ unm_idx,
                                const float* __restrict__ sums,
                                const float* __restrict__ counts,
                                float* __restrict__ xm)
{
    size_t t = (size_t)blockIdx.x * blockDim.x + threadIdx.x;
    if (t >= (size_t)BB * ND * CC) return;
    int c = (int)(t % CC);
    size_t r = t / CC;
    int i = (int)(r % ND), b = (int)(r / ND);
    // unm rows [0,1024)
    xm[((size_t)b * NM + i) * CC + c] =
        xln[((size_t)b * NN + a_idx[unm_idx[(size_t)b * ND + i]]) * CC + c];
    // dst rows [1024,2048)
    float d0 = xln[((size_t)b * NN + b_idx[i]) * CC + c];
    float sm = sums[((size_t)b * ND + i) * CC + c];
    float ct = counts[(size_t)b * ND + i];
    xm[((size_t)b * NM + ND + i) * CC + c] = (d0 + sm) / (1.0f + ct);
}

// Online-softmax attention: wave per (b, h, q). Q:(B,Nq,C) K/V:(B,Nk,C)
__global__ void __launch_bounds__(256)
attn_kernel(const float* __restrict__ Q, const float* __restrict__ Km,
            const float* __restrict__ Vm, float* __restrict__ Out,
            int Nq, int Nk)
{
    int gw   = (int)((blockIdx.x * (size_t)blockDim.x + threadIdx.x) >> 5);
    int lane = threadIdx.x & 31;
    if (gw >= BB * HH * Nq) return;
    int q = gw % Nq; int bh = gw / Nq; int h = bh % HH; int b = bh / HH;
    const float scale = 0.15811388300841898f; // 1/sqrt(40)

    const float* qp = Q + ((size_t)b * Nq + q) * CC + h * DHH;
    float qv[DHH];
#pragma unroll
    for (int d = 0; d < DHH; ++d) qv[d] = qp[d];

    float m = -1e30f, l = 0.f, o[DHH];
#pragma unroll
    for (int d = 0; d < DHH; ++d) o[d] = 0.f;

    for (int j = lane; j < Nk; j += 32) {
        const float* kp = Km + ((size_t)b * Nk + j) * CC + h * DHH;
        float s = 0.f;
#pragma unroll
        for (int d = 0; d < DHH; ++d) s += qv[d] * kp[d];
        s *= scale;
        float mn = fmaxf(m, s);
        float cO = __expf(m - mn);
        float p  = __expf(s - mn);
        const float* vp = Vm + ((size_t)b * Nk + j) * CC + h * DHH;
        l = l * cO + p;
#pragma unroll
        for (int d = 0; d < DHH; ++d) o[d] = o[d] * cO + p * vp[d];
        m = mn;
    }
    for (int off = 16; off; off >>= 1) {
        float m2 = __shfl_xor(m, off, 32);
        float l2 = __shfl_xor(l, off, 32);
        float mn = fmaxf(m, m2);
        float c1 = __expf(m - mn), c2 = __expf(m2 - mn);
        l = l * c1 + l2 * c2;
#pragma unroll
        for (int d = 0; d < DHH; ++d) {
            float o2 = __shfl_xor(o[d], off, 32);
            o[d] = o[d] * c1 + o2 * c2;
        }
        m = mn;
    }
    if (lane == 0) {
        float inv = 1.f / l;
        float* op = Out + ((size_t)b * Nq + q) * CC + h * DHH;
#pragma unroll
        for (int d = 0; d < DHH; ++d) op[d] = o[d] * inv;
    }
}

// Build token -> merged-row scatter map for unmerge
__global__ void build_map_kernel(const int* __restrict__ a_idx,
                                 const int* __restrict__ b_idx,
                                 const int* __restrict__ unm_idx,
                                 const int* __restrict__ src_idx,
                                 const int* __restrict__ dst_idx,
                                 int* __restrict__ map)
{
    int t = blockIdx.x * blockDim.x + threadIdx.x;
    if (t >= BB * NN) return;
    int b = t / NN, u = t % NN;
    if (u < ND) {
        map[(size_t)b * NN + b_idx[u]] = ND + u;
    } else if (u < 2 * ND) {
        int i = u - ND;
        map[(size_t)b * NN + a_idx[unm_idx[(size_t)b * ND + i]]] = i;
    } else {
        int k = u - 2 * ND;
        map[(size_t)b * NN + a_idx[src_idx[(size_t)b * RR + k]]] =
            ND + dst_idx[(size_t)b * RR + k];
    }
}

// net1 = net + unmerge(xa)   (xa: (B, NM, C), map: token -> merged row)
__global__ void unmerge_residual_kernel(const float* __restrict__ net,
                                        const float* __restrict__ xa,
                                        const int* __restrict__ map,
                                        float* __restrict__ out)
{
    size_t t = (size_t)blockIdx.x * blockDim.x + threadIdx.x;
    if (t >= (size_t)BB * NN * CC) return;
    int c = (int)(t % CC);
    size_t r = t / CC;
    int tok = (int)(r % NN), b = (int)(r / NN);
    out[t] = net[t] + xa[((size_t)b * NM + map[(size_t)b * NN + tok]) * CC + c];
}

// GEGLU: act[m,n] = h1[m,n] * gelu_exact(h1[m, DFF+n]) -> f16
__global__ void geglu_kernel(const float* __restrict__ h1,
                             _Float16* __restrict__ act, size_t rows)
{
    size_t t = (size_t)blockIdx.x * blockDim.x + threadIdx.x;
    if (t >= rows * DFF) return;
    size_t mrow = t / DFF;
    int n = (int)(t % DFF);
    float xh = h1[mrow * NFF1 + n];
    float gv = h1[mrow * NFF1 + DFF + n];
    float ge = 0.5f * gv * (1.f + erff(gv * 0.70710678118654752f));
    act[t] = (_Float16)(xh * ge);
}

// ---------------------------------------------------------------------------
// Host: orchestrate the whole block on `stream`
// ---------------------------------------------------------------------------
extern "C" void kernel_launch(void* const* d_in, const int* in_sizes, int n_in,
                              void* d_out, int out_size, void* d_ws, size_t ws_size,
                              hipStream_t stream)
{
    (void)in_sizes; (void)n_in; (void)out_size;

    const float* net      = (const float*)d_in[0];
    const float* context  = (const float*)d_in[1];
    const float* ln1_g    = (const float*)d_in[2];
    const float* ln1_b    = (const float*)d_in[3];
    const float* ln2_g    = (const float*)d_in[4];
    const float* ln2_b    = (const float*)d_in[5];
    const float* ln3_g    = (const float*)d_in[6];
    const float* ln3_b    = (const float*)d_in[7];
    const float* a1_wq    = (const float*)d_in[8];
    const float* a1_wk    = (const float*)d_in[9];
    const float* a1_wv    = (const float*)d_in[10];
    const float* a1_wo    = (const float*)d_in[11];
    const float* a1_bo    = (const float*)d_in[12];
    const float* a2_wq    = (const float*)d_in[13];
    const float* a2_wk    = (const float*)d_in[14];
    const float* a2_wv    = (const float*)d_in[15];
    const float* a2_wo    = (const float*)d_in[16];
    const float* a2_bo    = (const float*)d_in[17];
    const float* ff_w1    = (const float*)d_in[18];
    const float* ff_b1    = (const float*)d_in[19];
    const float* ff_w2    = (const float*)d_in[20];
    const float* ff_b2    = (const float*)d_in[21];
    float* out            = (float*)d_out;

    // ---- workspace bump allocator (256B aligned) ----
    char* base = (char*)d_ws;
    size_t off = 0;
    auto alloc = [&](size_t bytes) -> void* {
        off = (off + 255) & ~(size_t)255;
        void* p = base + off;
        off += bytes;
        return p;
    };

    int*      a_idx   = (int*)alloc(NA * 4);
    int*      b_idx   = (int*)alloc(ND * 4);
    _Float16* wq1t    = (_Float16*)alloc((size_t)CC * CC * 2);
    _Float16* wk1t    = (_Float16*)alloc((size_t)CC * CC * 2);
    _Float16* wv1t    = (_Float16*)alloc((size_t)CC * CC * 2);
    _Float16* wo1t    = (_Float16*)alloc((size_t)CC * CC * 2);
    _Float16* wq2t    = (_Float16*)alloc((size_t)CC * CC * 2);
    _Float16* wk2t    = (_Float16*)alloc((size_t)CC * DC * 2);
    _Float16* wv2t    = (_Float16*)alloc((size_t)CC * DC * 2);
    _Float16* wo2t    = (_Float16*)alloc((size_t)CC * CC * 2);
    _Float16* ff1t    = (_Float16*)alloc((size_t)NFF1 * CC * 2);
    _Float16* ff2t    = (_Float16*)alloc((size_t)CC * DFF * 2);
    _Float16* a_n     = (_Float16*)alloc((size_t)BB * NA * CC * 2);
    _Float16* b_n     = (_Float16*)alloc((size_t)BB * ND * CC * 2);
    float*    scores  = (float*)alloc((size_t)BB * NA * ND * 4);
    float*    node_max= (float*)alloc((size_t)BB * NA * 4);
    int*      node_idx= (int*)alloc((size_t)BB * NA * 4);
    int*      edge_idx= (int*)alloc((size_t)BB * NA * 4);
    int*      src_idx = (int*)alloc((size_t)BB * RR * 4);
    int*      unm_idx = (int*)alloc((size_t)BB * ND * 4);
    int*      dst_idx = (int*)alloc((size_t)BB * RR * 4);
    float*    x_ln1   = (float*)alloc((size_t)BB * NN * CC * 4);
    float*    sums    = (float*)alloc((size_t)BB * ND * CC * 4);
    float*    counts  = (float*)alloc((size_t)BB * ND * 4);
    float*    xm      = (float*)alloc((size_t)BB * NM * CC * 4);
    _Float16* xm16    = (_Float16*)alloc((size_t)BB * NM * CC * 2);
    float*    q1      = (float*)alloc((size_t)BB * NM * CC * 4);
    float*    k1      = (float*)alloc((size_t)BB * NM * CC * 4);
    float*    v1      = (float*)alloc((size_t)BB * NM * CC * 4);
    float*    o1      = (float*)alloc((size_t)BB * NM * CC * 4);
    _Float16* o1_16   = (_Float16*)alloc((size_t)BB * NM * CC * 2);
    float*    xa      = (float*)alloc((size_t)BB * NM * CC * 4);
    int*      map     = (int*)alloc((size_t)BB * NN * 4);
    float*    net1    = (float*)alloc((size_t)BB * NN * CC * 4);
    float*    x2      = (float*)alloc((size_t)BB * NN * CC * 4);
    _Float16* x2_16   = (_Float16*)alloc((size_t)BB * NN * CC * 2);
    float*    q2      = (float*)alloc((size_t)BB * NN * CC * 4);
    _Float16* ctx16   = (_Float16*)alloc((size_t)LCP * DC * 2);     // zero-padded to 160 rows
    float*    k2      = (float*)alloc((size_t)LCP * CC * 4);        // rows 0..153 valid
    float*    v2      = (float*)alloc((size_t)LCP * CC * 4);
    float*    o2      = (float*)alloc((size_t)BB * NN * CC * 4);
    _Float16* o2_16   = (_Float16*)alloc((size_t)BB * NN * CC * 2);
    float*    net2    = (float*)alloc((size_t)BB * NN * CC * 4);
    float*    x3      = (float*)alloc((size_t)BB * NN * CC * 4);
    _Float16* x3_16   = (_Float16*)alloc((size_t)BB * NN * CC * 2);
    float*    h1      = (float*)alloc((size_t)BB * NN * NFF1 * 4);
    _Float16* act16   = (_Float16*)alloc((size_t)BB * NN * DFF * 2);
    if (off > ws_size) return;  // workspace too small: well-defined no-op

    auto blocks = [](size_t threads) {
        return dim3((unsigned)((threads + 255) / 256));
    };
    auto wave_blocks = [](size_t waves) {
        return dim3((unsigned)((waves * 32 + 255) / 256));
    };
    auto gemm = [&](const _Float16* A, const _Float16* Bt, float* Cp,
                    const float* bias, const float* resid, int M, int N, int K) {
        int tiles = (M / 16) * (N / 32);   // M%16==0, N%32==0 guaranteed by construction
        wmma_gemm_kernel<<<dim3((tiles + 7) / 8), dim3(256), 0, stream>>>(
            A, Bt, Cp, bias, resid, M, N, K);
    };
    auto transp = [&](const float* src, _Float16* dst, int K, int N) {
        transpose_f32_f16_kernel<<<blocks((size_t)K * N), 256, 0, stream>>>(src, dst, K, N);
    };

    // 0) fixed ToMe partition indices
    build_idx_kernel<<<blocks(NN), 256, 0, stream>>>(a_idx, b_idx);

    // 1) weight transpose + f16 conversion
    transp(a1_wq, wq1t, CC, CC);   transp(a1_wk, wk1t, CC, CC);
    transp(a1_wv, wv1t, CC, CC);   transp(a1_wo, wo1t, CC, CC);
    transp(a2_wq, wq2t, CC, CC);   transp(a2_wk, wk2t, DC, CC);
    transp(a2_wv, wv2t, DC, CC);   transp(a2_wo, wo2t, CC, CC);
    transp(ff_w1, ff1t, CC, NFF1); transp(ff_w2, ff2t, DFF, CC);

    // 2) normalized metric gathers
    normalize_gather_kernel<<<wave_blocks((size_t)BB * NA), 256, 0, stream>>>(net, a_idx, a_n, NA);
    normalize_gather_kernel<<<wave_blocks((size_t)BB * ND), 256, 0, stream>>>(net, b_idx, b_n, ND);

    // 3) scores = a_n @ b_n^T per batch (WMMA)
    for (int b = 0; b < BB; ++b)
        gemm(a_n + (size_t)b * NA * CC, b_n + (size_t)b * ND * CC,
             scores + (size_t)b * NA * ND, nullptr, nullptr, NA, ND, CC);

    // 4) node max/argmax, stable descending rank, index split
    rowmax_argmax_kernel<<<wave_blocks((size_t)BB * NA), 256, 0, stream>>>(scores, node_max, node_idx);
    rank_kernel<<<blocks((size_t)BB * NA), 256, 0, stream>>>(node_max, edge_idx);
    split_idx_kernel<<<blocks((size_t)BB * NA), 256, 0, stream>>>(edge_idx, node_idx, src_idx, unm_idx, dst_idx);

    // 5) LN1 + merge
    ln_kernel<<<wave_blocks((size_t)BB * NN), 256, 0, stream>>>(net, ln1_g, ln1_b, x_ln1, BB * NN);
    hipMemsetAsync(sums,   0, (size_t)BB * ND * CC * 4, stream);
    hipMemsetAsync(counts, 0, (size_t)BB * ND * 4, stream);
    counts_kernel<<<blocks((size_t)BB * RR), 256, 0, stream>>>(dst_idx, counts);
    merge_sums_kernel<<<blocks((size_t)BB * RR * CC), 256, 0, stream>>>(x_ln1, a_idx, src_idx, dst_idx, sums);
    build_xm_kernel<<<blocks((size_t)BB * ND * CC), 256, 0, stream>>>(x_ln1, a_idx, b_idx, unm_idx, sums, counts, xm);

    // 6) attn1 (self) on merged tokens
    f32_to_f16_kernel<<<blocks((size_t)BB * NM * CC), 256, 0, stream>>>(xm, xm16, (size_t)BB * NM * CC);
    gemm(xm16, wq1t, q1, nullptr, nullptr, BB * NM, CC, CC);
    gemm(xm16, wk1t, k1, nullptr, nullptr, BB * NM, CC, CC);
    gemm(xm16, wv1t, v1, nullptr, nullptr, BB * NM, CC, CC);
    attn_kernel<<<wave_blocks((size_t)BB * HH * NM), 256, 0, stream>>>(q1, k1, v1, o1, NM, NM);
    f32_to_f16_kernel<<<blocks((size_t)BB * NM * CC), 256, 0, stream>>>(o1, o1_16, (size_t)BB * NM * CC);
    gemm(o1_16, wo1t, xa, a1_bo, nullptr, BB * NM, CC, CC);

    // 7) unmerge + residual
    build_map_kernel<<<blocks((size_t)BB * NN), 256, 0, stream>>>(a_idx, b_idx, unm_idx, src_idx, dst_idx, map);
    unmerge_residual_kernel<<<blocks((size_t)BB * NN * CC), 256, 0, stream>>>(net, xa, map, net1);

    // 8) attn2 (cross) with context; context rows zero-padded 154 -> 160
    ln_kernel<<<wave_blocks((size_t)BB * NN), 256, 0, stream>>>(net1, ln2_g, ln2_b, x2, BB * NN);
    f32_to_f16_kernel<<<blocks((size_t)BB * NN * CC), 256, 0, stream>>>(x2, x2_16, (size_t)BB * NN * CC);
    gemm(x2_16, wq2t, q2, nullptr, nullptr, BB * NN, CC, CC);
    hipMemsetAsync(ctx16, 0, (size_t)LCP * DC * 2, stream);
    f32_to_f16_kernel<<<blocks((size_t)BB * LC * DC), 256, 0, stream>>>(context, ctx16, (size_t)BB * LC * DC);
    gemm(ctx16, wk2t, k2, nullptr, nullptr, LCP, CC, DC);
    gemm(ctx16, wv2t, v2, nullptr, nullptr, LCP, CC, DC);
    attn_kernel<<<wave_blocks((size_t)BB * HH * NN), 256, 0, stream>>>(q2, k2, v2, o2, NN, LC);
    f32_to_f16_kernel<<<blocks((size_t)BB * NN * CC), 256, 0, stream>>>(o2, o2_16, (size_t)BB * NN * CC);
    gemm(o2_16, wo2t, net2, a2_bo, net1, BB * NN, CC, CC);   // epilogue: +bias +residual

    // 9) GEGLU feed-forward with fused residual into d_out
    ln_kernel<<<wave_blocks((size_t)BB * NN), 256, 0, stream>>>(net2, ln3_g, ln3_b, x3, BB * NN);
    f32_to_f16_kernel<<<blocks((size_t)BB * NN * CC), 256, 0, stream>>>(x3, x3_16, (size_t)BB * NN * CC);
    gemm(x3_16, ff1t, h1, ff_b1, nullptr, BB * NN, NFF1, CC);
    geglu_kernel<<<blocks((size_t)BB * NN * DFF), 256, 0, stream>>>(h1, act16, (size_t)BB * NN);
    gemm(act16, ff2t, out, ff_b2, net2, BB * NN, CC, DFF);
}
